// MultKAN_43456479101141
// MI455X (gfx1250) — matrix operand order repaired and checked by reading, not compiled, not run on hardware
//
#include <hip/hip_runtime.h>
#include <hip/hip_bf16.h>

typedef __bf16 bf16;
typedef __attribute__((ext_vector_type(16))) __bf16 v16bf;
typedef __attribute__((ext_vector_type(8)))  __bf16 v8bf;
typedef __attribute__((ext_vector_type(8)))  float  v8f;

#define TM 128
#define TN 128
#define KC 64
#define AST 72              // padded LDS row stride (bf16 elems): conflict-free frags
#define WST 72
#define ABUF (TM * AST)     // 9216 elems = 18 KB
#define WBUF (TN * WST)     // 18 KB
#define NTHR 256

// Direct cubic uniform B-spline: knots t_j = 0.4j - 2.2, G=8 bases.
// Only 4 bases nonzero at x: j = k-3..k, k = floor((x+2.2)/0.4).
// Equals the reference's truncated Cox-de-Boor recursion (OOB j dropped).
__device__ __forceinline__ void bspline_taps(float x, float t[4], int& j0) {
    float s = 2.5f * x + 5.5f;
    s = fminf(fmaxf(s, -1.0f), 12.0f);
    float fk = floorf(s);
    int   k  = (int)fk;
    float u  = s - fk;
    float um = 1.0f - u;
    float u2 = u * u, u3 = u2 * u;
    const float c6 = 0.16666667f;
    t[0] = um * um * um * c6;
    t[1] = (3.0f * u3 - 6.0f * u2 + 4.0f) * c6;
    t[2] = (-3.0f * u3 + 3.0f * u2 + 3.0f * u + 1.0f) * c6;
    t[3] = u3 * c6;
    j0 = k - 3;
}

// KC-deep WMMA sweep. A: (TM x KC) bf16 @ stride AST. W: (TN x KC) = B^T @ WST.
// Wave tile 64x32 = 4x2 fragments, f32 accumulate.
__device__ __forceinline__ void wmma_sweep(const bf16* __restrict__ Ab,
                                           const bf16* __restrict__ Wb,
                                           v8f acc[4][2],
                                           int wm, int wn, int lh, int lm) {
#pragma unroll
    for (int kk = 0; kk < KC; kk += 32) {
        v16bf a[4], b[2];
#pragma unroll
        for (int f = 0; f < 4; ++f) {           // A frag per ISA 7.12.2
            int row = wm + 16 * f + lm;
            int kb  = kk + lh * 8;
            v8bf lo = *(const v8bf*)(Ab + row * AST + kb);
            v8bf hi = *(const v8bf*)(Ab + row * AST + kb + 16);
#pragma unroll
            for (int e = 0; e < 8; ++e) { a[f][e] = lo[e]; a[f][8 + e] = hi[e]; }
        }
#pragma unroll
        for (int f = 0; f < 2; ++f) {           // B frag: 2x16B (16B-aligned rows)
            int col = wn + 16 * f + lm;
            int kb  = kk + lh * 8;
            v8bf blo = *(const v8bf*)(Wb + col * WST + kb);
            v8bf bhi = *(const v8bf*)(Wb + col * WST + kb + 16);
#pragma unroll
            for (int e = 0; e < 8; ++e) { b[f][e] = blo[e]; b[f][8 + e] = bhi[e]; }
        }
#pragma unroll
        for (int i = 0; i < 4; ++i)
#pragma unroll
            for (int j = 0; j < 2; ++j)
                acc[i][j] = __builtin_amdgcn_wmma_f32_16x16x32_bf16(
                    false, a[i], false, b[j], (short)0, acc[i][j], false, false);
    }
}

__global__ __launch_bounds__(NTHR) void kan_layer(
    const float* __restrict__ X,      // (4096, din)
    const float* __restrict__ coef,   // (din, dout, 8)
    const float* __restrict__ sb,     // (din, dout)
    const float* __restrict__ sp,     // (din, dout)
    const float* __restrict__ sub_s, const float* __restrict__ sub_b,   // (dout)
    const float* __restrict__ node_s, const float* __restrict__ node_b, // (dnext)
    float* __restrict__ Y,            // (4096, dnext)
    int din, int dout, int n_add, int dnext)
{
    __shared__ __align__(16) bf16 Alds[2 * ABUF];   // 36 KB ping-pong
    __shared__ __align__(16) bf16 Wlds[2 * WBUF];   // 36 KB ping-pong

    const int tid  = threadIdx.x;
    const int b0   = blockIdx.x * TM;   // fast axis -> W panel stays in L2
    const int c0   = blockIdx.y * TN;
    const int lane = tid & 31;
    const int wid  = tid >> 5;
    const int wm   = (wid >> 2) * 64;
    const int wn   = (wid & 3) * 32;
    const int lh   = lane >> 4;
    const int lm   = lane & 15;

    v8f acc[4][2];
#pragma unroll
    for (int i = 0; i < 4; ++i)
#pragma unroll
        for (int j = 0; j < 2; ++j)
#pragma unroll
            for (int e = 0; e < 8; ++e) acc[i][j][e] = 0.0f;

    // ======== Pass 1: spline GEMM, register-prefetch + ping-pong LDS ========
    const int S1 = din / 8;
    float  xr[4]; float4 w0r[4], w1r[4]; float spr[4];

    // prologue: load stage 0
#pragma unroll
    for (int q = 0; q < 4; ++q) {
        int p = tid + NTHR * q;                      // 128*8
        int m = p >> 3, ii = p & 7;
        xr[q] = X[(size_t)(b0 + m) * din + ii];
    }
#pragma unroll
    for (int q = 0; q < 4; ++q) {
        int p = tid + NTHR * q;                      // 8*128
        int i = p >> 7, o = p & 127;
        size_t io = (size_t)i * dout + (c0 + o);
        const float* cp = coef + io * 8;
        w0r[q] = *(const float4*)cp;
        w1r[q] = *(const float4*)(cp + 4);
        spr[q] = sp[io];
    }

    for (int s = 0; s < S1; ++s) {
        bf16* Ab = Alds + (s & 1) * ABUF;
        bf16* Wb = Wlds + (s & 1) * WBUF;

        // convert current stage registers -> LDS
#pragma unroll
        for (int q = 0; q < 4; ++q) {
            int p = tid + NTHR * q;
            int m = p >> 3, ii = p & 7;
            float tp[4]; int j0;
            bspline_taps(xr[q], tp, j0);
            bf16* dst = Ab + m * AST + ii * 8;
            v8bf z;
#pragma unroll
            for (int e = 0; e < 8; ++e) z[e] = (bf16)0.0f;
            *(v8bf*)dst = z;                         // zero 16B slot
#pragma unroll
            for (int d = 0; d < 4; ++d) {            // scatter 4 taps (same-
                int j = j0 + d;                      // thread DS ops in order)
                if ((unsigned)j < 8u) dst[j] = (bf16)tp[d];
            }
        }
#pragma unroll
        for (int q = 0; q < 4; ++q) {
            int p = tid + NTHR * q;
            int i = p >> 7, o = p & 127;
            v8bf wb;
            wb[0] = (bf16)(w0r[q].x * spr[q]); wb[1] = (bf16)(w0r[q].y * spr[q]);
            wb[2] = (bf16)(w0r[q].z * spr[q]); wb[3] = (bf16)(w0r[q].w * spr[q]);
            wb[4] = (bf16)(w1r[q].x * spr[q]); wb[5] = (bf16)(w1r[q].y * spr[q]);
            wb[6] = (bf16)(w1r[q].z * spr[q]); wb[7] = (bf16)(w1r[q].w * spr[q]);
            *(v8bf*)(Wb + o * WST + i * 8) = wb;
        }

        // issue next stage's global loads before the barrier: their latency
        // hides behind this stage's 16 WMMAs (waits land at next convert).
        if (s + 1 < S1) {
            int i0n = (s + 1) * 8;
#pragma unroll
            for (int q = 0; q < 4; ++q) {
                int p = tid + NTHR * q;
                int m = p >> 3, ii = p & 7;
                xr[q] = X[(size_t)(b0 + m) * din + (i0n + ii)];
            }
#pragma unroll
            for (int q = 0; q < 4; ++q) {
                int p = tid + NTHR * q;
                int i = p >> 7, o = p & 127;
                size_t io = (size_t)(i0n + i) * dout + (c0 + o);
                const float* cp = coef + io * 8;
                w0r[q] = *(const float4*)cp;
                w1r[q] = *(const float4*)(cp + 4);
                spr[q] = sp[io];
            }
        }

        // single barrier per stage: __syncthreads lowers to
        // s_wait_dscnt 0 + s_barrier, so every wave's sweep loads from the
        // other buffer have retired before anyone overwrites it next stage.
        __syncthreads();
        wmma_sweep(Ab, Wb, acc, wm, wn, lh, lm);
    }
    __syncthreads();

    // ======== Pass 2: silu(x) @ sb, ping-pong LDS ========
    const int S2 = din / KC;
    for (int s = 0; s < S2; ++s) {
        bf16* Ab = Alds + (s & 1) * ABUF;
        bf16* Wb = Wlds + (s & 1) * WBUF;
        int i1 = s * KC;
        for (int e = 0; e < 32; ++e) {
            int p = tid + NTHR * e;                  // 128*64
            int m = p >> 6, ii = p & 63;
            float xv = X[(size_t)(b0 + m) * din + (i1 + ii)];
            Ab[m * AST + ii] = (bf16)(xv / (1.0f + __expf(-xv)));
        }
        for (int e = 0; e < 32; ++e) {
            int p = tid + NTHR * e;                  // 64*128
            int o = p & 127, ii = p >> 7;
            Wb[o * WST + ii] = (bf16)sb[(size_t)(i1 + ii) * dout + (c0 + o)];
        }
        __syncthreads();
        wmma_sweep(Ab, Wb, acc, wm, wn, lh, lm);
    }
    __syncthreads();

    // ======== Epilogue: affine params via LDS, pair-mult via shfl_xor ========
    float* subs = (float*)Alds;        // reuse LDS (all sweeps retired)
    float* subb = subs + TN;
    float* nds  = subb + TN;
    float* ndb  = nds + TN;
    const bool addTile = (c0 < n_add);
    const int  obase   = n_add + ((c0 - n_add) >> 1);
    if (tid < TN) { subs[tid] = sub_s[c0 + tid]; subb[tid] = sub_b[c0 + tid]; }
    if (addTile) {
        if (tid < TN) { nds[tid] = node_s[c0 + tid]; ndb[tid] = node_b[c0 + tid]; }
    } else {
        if (tid < TN / 2) { nds[tid] = node_s[obase + tid]; ndb[tid] = node_b[obase + tid]; }
    }
    __syncthreads();

#pragma unroll
    for (int i = 0; i < 4; ++i)
#pragma unroll
        for (int j = 0; j < 2; ++j) {
            int n = wn + 16 * j + lm;               // C layout: N = lane%16
#pragma unroll
            for (int r = 0; r < 8; ++r) {
                int m = wm + 16 * i + lh * 8 + r;   // VGPR r -> M = r / r+8
                float y = subs[n] * acc[i][j][r] + subb[n];
                if (addTile) {
                    Y[(size_t)(b0 + m) * dnext + (c0 + n)] = nds[n] * y + ndb[n];
                } else {
                    float py = __shfl_xor(y, 1, 32);     // partner column (wave32)
                    if ((lm & 1) == 0) {
                        int jj = n >> 1;
                        Y[(size_t)(b0 + m) * dnext + (obase + jj)] =
                            nds[jj] * (y * py) + ndb[jj];
                    }
                }
            }
        }
}

extern "C" void kernel_launch(void* const* d_in, const int* in_sizes, int n_in,
                              void* d_out, int out_size, void* d_ws, size_t ws_size,
                              hipStream_t stream)
{
    (void)in_sizes; (void)n_in; (void)out_size; (void)ws_size;
    const float* x = (const float*)d_in[0];

    const float* coef[3]; const float* sb[3]; const float* sp[3];
    const float* sus[3];  const float* sub[3];
    const float* nos[3];  const float* nob[3];
    for (int l = 0; l < 3; ++l) {
        int base = 1 + 7 * l;
        coef[l] = (const float*)d_in[base + 0];
        sb[l]   = (const float*)d_in[base + 1];
        sp[l]   = (const float*)d_in[base + 2];
        sus[l]  = (const float*)d_in[base + 3];
        sub[l]  = (const float*)d_in[base + 4];
        nos[l]  = (const float*)d_in[base + 5];
        nob[l]  = (const float*)d_in[base + 6];
    }

    float* act1 = (float*)d_ws;                 // 4096 x 2560 f32 (40 MB)
    float* act2 = act1 + (size_t)4096 * 2560;   // 4096 x 2560 f32 (40 MB)
    float* out  = (float*)d_out;                // 4096 x 1024 f32

    dim3 blk(NTHR);
    kan_layer<<<dim3(4096 / TM, 3072 / TN), blk, 0, stream>>>(
        x, coef[0], sb[0], sp[0], sus[0], sub[0], nos[0], nob[0],
        act1, 1024, 3072, 2048, 2560);
    kan_layer<<<dim3(4096 / TM, 3072 / TN), blk, 0, stream>>>(
        act1, coef[1], sb[1], sp[1], sus[1], sub[1], nos[1], nob[1],
        act2, 2560, 3072, 2048, 2560);
    kan_layer<<<dim3(4096 / TM, 1024 / TN), blk, 0, stream>>>(
        act2, coef[2], sb[2], sp[2], sus[2], sub[2], nos[2], nob[2],
        out, 2560, 1024, 1024, 1024);
}
